// PointNet2SemSeg_34170759807383
// MI455X (gfx1250) — matrix operand order repaired and checked by reading, not compiled, not run on hardware
//
#include <hip/hip_runtime.h>
#include <cstddef>
#include <cstdint>

#define BATCH 8
#define NPTS0 16384
#define NS 32
#define NUM_CLASSES 13

typedef __bf16 bf16_t;
typedef __attribute__((ext_vector_type(16))) __bf16 v16bf;
typedef __attribute__((ext_vector_type(8)))  __bf16 v8bf;
typedef __attribute__((ext_vector_type(8)))  float  v8f;
typedef __attribute__((ext_vector_type(4)))  unsigned int u32x4;
typedef __attribute__((ext_vector_type(8)))  int i32x8;
typedef __attribute__((ext_vector_type(4)))  int i32x4;

#if defined(__has_builtin)
# if __has_builtin(__builtin_amdgcn_tensor_load_to_lds)
#  define USE_TDM 1
# endif
#endif
#ifndef USE_TDM
# define USE_TDM 0
#endif

static __device__ __forceinline__ bf16_t f2bf(float f) {
  union { float f; unsigned u; } x; x.f = f;
  unsigned r = x.u + 0x7fffu + ((x.u >> 16) & 1u);   // round-to-nearest-even
  return __builtin_bit_cast(bf16_t, (unsigned short)(r >> 16));
}
static __device__ __forceinline__ float bf2f(bf16_t b) {
  unsigned short h = __builtin_bit_cast(unsigned short, b);
  union { unsigned u; float f; } x; x.u = ((unsigned)h) << 16;
  return x.f;
}

// ---------------- farthest point sampling: 1 workgroup per batch ----------------
__global__ void fps_kernel(const float* __restrict__ xyz, int Npts, int npoint,
                           int* __restrict__ fidx, float* __restrict__ dist) {
  const int b = blockIdx.x;
  const float* X = xyz + (size_t)b * Npts * 3;
  float* D = dist + (size_t)b * Npts;
  __shared__ float sval[512];
  __shared__ int   sidx[512];
  const int t = threadIdx.x, nt = blockDim.x;
  for (int i = t; i < Npts; i += nt) D[i] = 1e10f;
  __syncthreads();
  int far = 0;
  for (int it = 0; it < npoint; ++it) {
    if (t == 0) fidx[b * npoint + it] = far;
    const float cx = X[far*3+0], cy = X[far*3+1], cz = X[far*3+2];
    float best = -1.0f; int bi = 0;
    for (int i = t; i < Npts; i += nt) {
      float dx = X[i*3+0]-cx, dy = X[i*3+1]-cy, dz = X[i*3+2]-cz;
      float d = dx*dx + dy*dy + dz*dz;
      float dm = fminf(D[i], d);
      D[i] = dm;
      if (dm > best) { best = dm; bi = i; }
    }
    sval[t] = best; sidx[t] = bi;
    __syncthreads();
    for (int s = nt >> 1; s > 0; s >>= 1) {
      if (t < s && sval[t+s] > sval[t]) { sval[t] = sval[t+s]; sidx[t] = sidx[t+s]; }
      __syncthreads();
    }
    far = sidx[0];
    __syncthreads();
  }
}

__global__ void gather_xyz_kernel(const float* __restrict__ xyz, const int* __restrict__ fidx,
                                  int Npts, int npoint, float* __restrict__ newxyz, int total) {
  int e = blockIdx.x * blockDim.x + threadIdx.x;
  if (e >= total) return;
  int c = e % 3; int bm = e / 3; int b = bm / npoint;
  int pi = fidx[bm];
  newxyz[e] = xyz[((size_t)b * Npts + pi) * 3 + c];
}

// ---------------- ball query: first nsample indices within radius ----------------
__global__ void ballquery_kernel(const float* __restrict__ xyz, const float* __restrict__ newxyz,
                                 int Npts, int npoint, int nsample, float r2,
                                 int* __restrict__ gidx, int total) {
  int idx = blockIdx.x * blockDim.x + threadIdx.x;   // b*npoint + m
  if (idx >= total) return;
  const float* X = xyz + (size_t)(idx / npoint) * Npts * 3;
  const float cx = newxyz[(size_t)idx*3+0], cy = newxyz[(size_t)idx*3+1], cz = newxyz[(size_t)idx*3+2];
  int* out = gidx + (size_t)idx * nsample;
  int cnt = 0, first = 0;
  for (int i = 0; i < Npts && cnt < nsample; ++i) {
    float dx = X[i*3+0]-cx, dy = X[i*3+1]-cy, dz = X[i*3+2]-cz;
    float d2 = dx*dx + dy*dy + dz*dz;
    if (d2 <= r2) { if (cnt == 0) first = i; out[cnt++] = i; }
  }
  for (; cnt < nsample; ++cnt) out[cnt] = first;
}

// ---- build grouped feature rows [B*npoint*nsample, Kpad] bf16 (zero-padded) ----
__global__ void group_stage_kernel(const float* __restrict__ xyz, const float* __restrict__ newxyz,
                                   const float* __restrict__ points, const int* __restrict__ gidx,
                                   int Npts, int npoint, int nsample, int C, int Kpad,
                                   bf16_t* __restrict__ out, int total) {
  int e = blockIdx.x * blockDim.x + threadIdx.x;
  if (e >= total) return;
  int k   = e % Kpad;
  int row = e / Kpad;
  int bm  = row / nsample;
  int b   = bm / npoint;
  int pi  = gidx[row];
  float v = 0.0f;
  if (k < 3) {
    v = xyz[((size_t)b * Npts + pi) * 3 + k] - newxyz[(size_t)bm * 3 + k];
  } else if (points != nullptr && k < 3 + C) {
    v = points[((size_t)b * Npts + pi) * C + (k - 3)];
  }
  out[e] = f2bf(v);
}

// ---- weight prep: W f32 [K,Cout] -> Wt bf16 [Npad][Kpad] (transposed, padded); bias -> bpad f32 ----
__global__ void prep_w_kernel(const float* __restrict__ W, const float* __restrict__ bias,
                              int K, int Cout, int Kpad, int Npad,
                              bf16_t* __restrict__ Wt, float* __restrict__ bpad) {
  int e = blockIdx.x * blockDim.x + threadIdx.x;
  int total = Npad * Kpad;
  if (e < Npad) bpad[e] = (e < Cout) ? bias[e] : 0.0f;
  if (e >= total) return;
  int k = e % Kpad, n = e / Kpad;
  float v = (k < K && n < Cout) ? W[(size_t)k * Cout + n] : 0.0f;
  Wt[e] = f2bf(v);
}

// ---------------- WMMA GEMM: out = act(A[rows,Kpad] @ Wt^T + b) ----------------
// One block = one 16-wide N-tile x four 16-row M-tiles (one per wave).
// The block's B tile (16 x Kpad bf16, contiguous in Wt) is DMA'd into LDS with the
// Tensor Data Mover (tensor_load_to_lds + s_wait_tensorcnt), then all 4 waves feed
// their WMMA B-fragments from LDS while streaming A-fragments from global.
__global__ void __launch_bounds__(128)
wmma_gemm_kernel(const bf16_t* __restrict__ A, const bf16_t* __restrict__ Wt,
                 const float* __restrict__ bpad,
                 int rows, int Kpad, int Npad,
                 bf16_t* __restrict__ outBf, float* __restrict__ outF32,
                 int outStride, int relu) {
  __shared__ __align__(64) bf16_t sB[16 * 768];      // max tile: 16 x Kpad(<=768) bf16 = 24KB
  const int lane = threadIdx.x;        // 0..31
  const int wave = threadIdx.y;        // 0..3
  const int tilesN = Npad >> 4;
  const int tilesM = rows >> 4;
  const int nt = blockIdx.x % tilesN;
  const int mt = (blockIdx.x / tilesN) * 4 + wave;
  const int nbase = nt << 4;
  const bf16_t* gB = Wt + (size_t)nbase * Kpad;      // contiguous 16*Kpad elements
  const unsigned te = (unsigned)(Kpad << 4);         // tile elements (<= 12288)

#if USE_TDM
  if (wave == 0) {
    // ---- Tensor DMA descriptor (ISA cdna5 ch.8), 1-D tile of te bf16 elements ----
    unsigned long long ga = (unsigned long long)(uintptr_t)gB;
    unsigned la = (unsigned)(size_t)(void*)&sB[0];   // flat addr low 32b == LDS byte offset
    u32x4 g0 = { 1u,                                 // count=1, user mode
                 la,                                 // lds_addr [63:32]
                 (unsigned)ga,                       // global_addr [95:64]
                 (unsigned)(((ga >> 32) & 0x01ffffffu) | 0x80000000u) }; // addr[56:32] | type=2
    i32x8 g1 = { 0x00010000,                         // data_size=1 (2 bytes), mask=0
                 (int)((te & 0xffffu) << 16),        // tensor_dim0[15:0]
                 (int)(((te >> 16) & 0xffffu) | (1u << 16)),  // tensor_dim0[31:16] | tensor_dim1=1
                 (int)((te & 0xffffu) << 16),        // tensor_dim1 hi=0 | tile_dim0=te
                 1,                                  // tile_dim1=1, tile_dim2=0
                 (int)te,                            // tensor_dim0_stride[31:0]
                 0, 0 };                             // stride hi / tensor_dim1_stride = 0
    i32x4 gz = { 0, 0, 0, 0 };
#if defined(__clang_major__) && (__clang_major__ >= 23)
    i32x8 gz8 = { 0, 0, 0, 0, 0, 0, 0, 0 };
    __builtin_amdgcn_tensor_load_to_lds(g0, g1, gz, gz, gz8, 0);
#else
    __builtin_amdgcn_tensor_load_to_lds(g0, g1, gz, gz, 0);
#endif
    __builtin_amdgcn_s_wait_tensorcnt(0);
  }
  __syncthreads();
#else
  // fallback: cooperative vector copy into LDS
  {
    const int tid = wave * 32 + lane;
    const int nchunk = (int)(te >> 3);
    for (int i = tid; i < nchunk; i += 128)
      ((v8bf*)sB)[i] = ((const v8bf*)gB)[i];
  }
  __syncthreads();
#endif

  if (mt < tilesM) {                                 // wave-uniform guard
    const int mbase = mt << 4;
    const int hs = lane >> 4;          // half-select
    const int lr = lane & 15;
    const bf16_t* arow = A + (size_t)(mbase + lr) * Kpad;
    const bf16_t* srow = sB + (size_t)lr * Kpad + (size_t)hs * 16;
    v8f acc = {};
    for (int k0 = 0; k0 < Kpad; k0 += 32) {
      // A frag: lanes 0-15 hold K {0..15}; lanes 16-31 hold K {16..31}
      v8bf a0 = *(const v8bf*)(arow + k0 + hs * 16);
      v8bf a1 = *(const v8bf*)(arow + k0 + 8 + hs * 16);
      v16bf a;
      #pragma unroll
      for (int i = 0; i < 8; ++i) { a[i] = a0[i]; a[8 + i] = a1[i]; }
      // B frag from LDS: lane holds column n=lr, 16 contiguous K at k0 + hs*16
      v16bf bm = *(const v16bf*)(srow + k0);
      acc = __builtin_amdgcn_wmma_f32_16x16x32_bf16(false, a, false, bm,
                                                    (short)0, acc, false, false);
    }
    const float bb = bpad[nbase + lr];
    #pragma unroll
    for (int r = 0; r < 8; ++r) {
      float v = acc[r] + bb;
      if (relu) v = fmaxf(v, 0.0f);
      const int row = mbase + r + (hs << 3);         // D layout: M = r + 8*hs
      const size_t o = (size_t)row * outStride + nbase + lr;
      if (outBf) outBf[o] = f2bf(v);
      else       outF32[o] = v;
    }
  }
}

// ---------------- max-pool over nsample neighbors ----------------
__global__ void maxpool_kernel(const bf16_t* __restrict__ act, int nsample, int C,
                               float* __restrict__ out, int total) {
  int e = blockIdx.x * blockDim.x + threadIdx.x;
  if (e >= total) return;
  int c = e % C; int bm = e / C;
  const bf16_t* p = act + ((size_t)bm * nsample) * C + c;
  float mx = -1e30f;
  for (int s = 0; s < nsample; ++s) mx = fmaxf(mx, bf2f(p[(size_t)s * C]));
  out[e] = mx;
}

// ---- FP stage: 3-NN inverse-distance interp + concat -> staging bf16 [rows,Kpad] ----
__global__ void fp_interp_stage_kernel(const float* __restrict__ xyz1, const float* __restrict__ xyz2,
                                       const float* __restrict__ p1, const float* __restrict__ p2,
                                       int N1, int N2, int C1, int C2, int Kpad,
                                       bf16_t* __restrict__ out, int total) {
  int idx = blockIdx.x * blockDim.x + threadIdx.x;   // b*N1 + i
  if (idx >= total) return;
  int b = idx / N1;
  const float px = xyz1[(size_t)idx*3+0], py = xyz1[(size_t)idx*3+1], pz = xyz1[(size_t)idx*3+2];
  const float* X2 = xyz2 + (size_t)b * N2 * 3;
  float d0 = 1e30f, d1 = 1e30f, d2 = 1e30f; int i0 = 0, i1 = 0, i2 = 0;
  for (int j = 0; j < N2; ++j) {
    float dx = X2[j*3+0]-px, dy = X2[j*3+1]-py, dz = X2[j*3+2]-pz;
    float d = dx*dx + dy*dy + dz*dz;
    if (d < d0)      { d2=d1; i2=i1; d1=d0; i1=i0; d0=d; i0=j; }
    else if (d < d1) { d2=d1; i2=i1; d1=d; i1=j; }
    else if (d < d2) { d2=d; i2=j; }
  }
  float w0 = 1.0f/(d0+1e-8f), w1 = 1.0f/(d1+1e-8f), w2 = 1.0f/(d2+1e-8f);
  float ws = w0+w1+w2; w0/=ws; w1/=ws; w2/=ws;
  const float* q0 = p2 + ((size_t)b*N2 + i0)*C2;
  const float* q1 = p2 + ((size_t)b*N2 + i1)*C2;
  const float* q2 = p2 + ((size_t)b*N2 + i2)*C2;
  bf16_t* o = out + (size_t)idx * Kpad;
  const int c1 = (p1 != nullptr) ? C1 : 0;
  for (int k = 0; k < Kpad; ++k) {
    float v = 0.0f;
    if (k < c1)           v = p1[(size_t)idx * C1 + k];
    else if (k < c1 + C2) { int c = k - c1; v = w0*q0[c] + w1*q1[c] + w2*q2[c]; }
    o[k] = f2bf(v);
  }
}

// ---------------- log_softmax over 13 classes (logits stride 16) ----------------
__global__ void logsoftmax_kernel(const float* __restrict__ logits, float* __restrict__ out, int total) {
  int i = blockIdx.x * blockDim.x + threadIdx.x;
  if (i >= total) return;
  const float* l = logits + (size_t)i * 16;
  float mx = -1e30f;
  for (int c = 0; c < NUM_CLASSES; ++c) mx = fmaxf(mx, l[c]);
  float s = 0.0f;
  for (int c = 0; c < NUM_CLASSES; ++c) s += expf(l[c] - mx);
  float lse = mx + logf(s);
  for (int c = 0; c < NUM_CLASSES; ++c) out[(size_t)i * NUM_CLASSES + c] = l[c] - lse;
}

extern "C" void kernel_launch(void* const* d_in, const int* in_sizes, int n_in,
                              void* d_out, int out_size, void* d_ws, size_t ws_size,
                              hipStream_t stream) {
  (void)out_size; (void)ws_size;
  // ---------- parameter discovery (dict insertion order vs JAX sorted order) ----------
  const float *saW[4][3], *saB[4][3], *fpW[4][3] = {}, *fpB[4][3] = {}, *hW[2], *hB[2];
  const int fpNL[4] = {2, 2, 2, 3};
  const float* xyzIn;
  int p = 0;
  auto nx = [&]() { return (const float*)d_in[p++]; };
  if (in_sizes[0] == BATCH * NPTS0 * 3) {            // xyz first (insertion order)
    xyzIn = (const float*)d_in[0]; p = 1;
    for (int l = 0; l < 4; ++l) for (int j = 0; j < 3; ++j)       { saW[l][j]=nx(); saB[l][j]=nx(); }
    for (int l = 0; l < 4; ++l) for (int j = 0; j < fpNL[l]; ++j) { fpW[l][j]=nx(); fpB[l][j]=nx(); }
    hW[0]=nx(); hB[0]=nx(); hW[1]=nx(); hB[1]=nx();
  } else {                                           // params first (alphabetical: fp, head, sa), xyz last
    for (int l = 0; l < 4; ++l) for (int j = 0; j < fpNL[l]; ++j) { fpW[l][j]=nx(); fpB[l][j]=nx(); }
    hW[0]=nx(); hB[0]=nx(); hW[1]=nx(); hB[1]=nx();
    for (int l = 0; l < 4; ++l) for (int j = 0; j < 3; ++j)       { saW[l][j]=nx(); saB[l][j]=nx(); }
    xyzIn = (const float*)d_in[n_in - 1];
  }

  // ---------- workspace layout ----------
  char* base = (char*)d_ws; size_t off = 0;
  auto alloc = [&](size_t bytes) -> void* {
    void* r = base + off; off = (off + bytes + 255) & ~((size_t)255); return r;
  };
  const int sa_np[4]   = {1024, 256, 64, 16};
  const int sa_mlp[4][3] = {{32,32,64},{64,64,128},{128,128,256},{256,256,512}};
  const float sa_r[4]  = {0.1f, 0.2f, 0.4f, 0.8f};

  float* distbuf = (float*)alloc((size_t)BATCH * NPTS0 * 4);
  int*   fidx    = (int*)  alloc((size_t)BATCH * 1024 * 4);
  int*   gidx    = (int*)  alloc((size_t)BATCH * 1024 * NS * 4);
  float* lx[4]; float* lp[4];
  for (int l = 0; l < 4; ++l) {
    lx[l] = (float*)alloc((size_t)BATCH * sa_np[l] * 3 * 4);
    lp[l] = (float*)alloc((size_t)BATCH * sa_np[l] * sa_mlp[l][2] * 4);
  }
  float* f3 = (float*)alloc((size_t)BATCH * 64   * 256 * 4);
  float* f2 = (float*)alloc((size_t)BATCH * 256  * 256 * 4);
  float* f1 = (float*)alloc((size_t)BATCH * 1024 * 128 * 4);
  bf16_t* Wt   = (bf16_t*)alloc((size_t)256 * 768 * 2);
  float*  bpad = (float*) alloc((size_t)512 * 4);
  float*  logits = (float*)alloc((size_t)BATCH * NPTS0 * 16 * 4);
  const size_t ACT_ELEMS = (size_t)16777216;         // max of staging/activation footprints
  bf16_t* bufS = (bf16_t*)alloc(ACT_ELEMS * 2);
  bf16_t* actA = (bf16_t*)alloc(ACT_ELEMS * 2);
  bf16_t* actB = (bf16_t*)alloc(ACT_ELEMS * 2);

  auto nb = [](long long t) { return (unsigned int)((t + 255) / 256); };

  auto gemm = [&](const bf16_t* A, const float* W, const float* bias, int rows,
                  int K, int Kpad, int Cout, bf16_t* oBf, float* oF32, int oStride, int relu) {
    int Npad = (Cout + 15) & ~15;
    prep_w_kernel<<<nb((long long)Npad * Kpad), 256, 0, stream>>>(W, bias, K, Cout, Kpad, Npad, Wt, bpad);
    int tilesN = Npad / 16;
    int blocks = tilesN * ((rows / 16 + 3) / 4);
    wmma_gemm_kernel<<<blocks, dim3(32, 4), 0, stream>>>(A, Wt, bpad, rows, Kpad, Npad,
                                                         oBf, oF32, oStride, relu);
  };

  // ---------- SA stages ----------
  const float* curXyz = xyzIn; int curN = NPTS0;
  const float* curPts = nullptr; int curC = 0;
  for (int l = 0; l < 4; ++l) {
    const int np = sa_np[l];
    fps_kernel<<<BATCH, 512, 0, stream>>>(curXyz, curN, np, fidx, distbuf);
    gather_xyz_kernel<<<nb((long long)BATCH * np * 3), 256, 0, stream>>>(curXyz, fidx, curN, np, lx[l], BATCH * np * 3);
    ballquery_kernel<<<nb(BATCH * np), 256, 0, stream>>>(curXyz, lx[l], curN, np, NS,
                                                         sa_r[l] * sa_r[l], gidx, BATCH * np);
    const int K = 3 + curC, Kpad = (K + 31) & ~31;
    const int rows = BATCH * np * NS;
    const long long tote = (long long)rows * Kpad;
    group_stage_kernel<<<nb(tote), 256, 0, stream>>>(curXyz, lx[l], curPts, gidx, curN, np, NS, curC,
                                                     Kpad, bufS, (int)tote);
    const bf16_t* Ain = bufS; int kc = K, kp = Kpad;
    for (int j = 0; j < 3; ++j) {
      const int Cout = sa_mlp[l][j];
      bf16_t* dst = (Ain == actA) ? actB : actA;
      gemm(Ain, saW[l][j], saB[l][j], rows, kc, kp, Cout, dst, nullptr, Cout, 1);
      Ain = dst; kc = Cout; kp = Cout;
    }
    maxpool_kernel<<<nb((long long)BATCH * np * kc), 256, 0, stream>>>(Ain, NS, kc, lp[l], BATCH * np * kc);
    curXyz = lx[l]; curN = np; curPts = lp[l]; curC = kc;
  }

  // ---------- FP stages ----------
  struct FpGeo { const float* x1; int N1; const float* x2; int N2;
                 const float* p1; int C1; const float* p2; int C2; float* fout; };
  const FpGeo fg[4] = {
    { lx[2], 64,    lx[3], 16,   lp[2],   256, lp[3], 512, f3      },
    { lx[1], 256,   lx[2], 64,   lp[1],   128, f3,    256, f2      },
    { lx[0], 1024,  lx[1], 256,  lp[0],   64,  f2,    256, f1      },
    { xyzIn, NPTS0, lx[0], 1024, nullptr, 0,   f1,    128, nullptr },
  };
  const int fp_mlp[4][3] = {{256,256,0},{256,256,0},{256,128,0},{128,128,128}};
  const bf16_t* fp3act = nullptr;
  for (int s = 0; s < 4; ++s) {
    const int N1 = fg[s].N1, N2 = fg[s].N2, C1 = fg[s].C1, C2 = fg[s].C2;
    const int K = C1 + C2, Kpad = (K + 31) & ~31;
    const int rows = BATCH * N1;
    fp_interp_stage_kernel<<<nb(rows), 256, 0, stream>>>(fg[s].x1, fg[s].x2, fg[s].p1, fg[s].p2,
                                                         N1, N2, C1, C2, Kpad, bufS, rows);
    const bf16_t* Ain = bufS; int kc = K, kp = Kpad;
    for (int j = 0; j < fpNL[s]; ++j) {
      const int Cout = fp_mlp[s][j];
      const bool last = (j == fpNL[s] - 1);
      bf16_t* dst = (Ain == actA) ? actB : actA;
      if (last && s < 3) {
        gemm(Ain, fpW[s][j], fpB[s][j], rows, kc, kp, Cout, nullptr, fg[s].fout, Cout, 1);
      } else {
        gemm(Ain, fpW[s][j], fpB[s][j], rows, kc, kp, Cout, dst, nullptr, Cout, 1);
        Ain = dst;
      }
      kc = Cout; kp = Cout;
    }
    if (s == 3) fp3act = Ain;
  }

  // ---------- head + log_softmax ----------
  const int rows0 = BATCH * NPTS0;
  bf16_t* h1 = (fp3act == actA) ? actB : actA;
  gemm(fp3act, hW[0], hB[0], rows0, 128, 128, 128, h1, nullptr, 128, 1);
  gemm(h1,     hW[1], hB[1], rows0, 128, 128, NUM_CLASSES, nullptr, logits, 16, 0);
  logsoftmax_kernel<<<nb(rows0), 256, 0, stream>>>(logits, (float*)d_out, rows0);
}
